// EasySession_17635135717526
// MI455X (gfx1250) — compile-verified, matrix operand order checked by reading, not run on hardware
//
#include <hip/hip_runtime.h>
#include <hip/hip_bf16.h>

// ---------------------------------------------------------------------------
// Problem constants (from the reference)
// ---------------------------------------------------------------------------
#define OO   3
#define HH   8
#define DD   128
#define HD   1024          // H*D
#define VV   100000
#define BB   256
#define SLEN 20
#define NN   (BB * SLEN)   // 5120
#define EE   (2 * NN)      // 10240

typedef __attribute__((ext_vector_type(16))) _Float16 v16h;
typedef __attribute__((ext_vector_type(8)))  _Float16 v8h;
typedef __attribute__((ext_vector_type(8)))  float    v8f;

// ---------------------------------------------------------------------------
// WMMA fragment loader. 16-bit A-matrix 16x32 layout (CDNA5 ISA 7.12.2):
//   lanes 0-15  : VGPR0..3 hold K=0..7,  VGPR4..7 hold K=16..23
//   lanes 16-31 : VGPR0..3 hold K=8..15, VGPR4..7 hold K=24..31
// => per lane two contiguous 8-half chunks at kBase+8*half and kBase+16+8*half.
// B (32x16, column-major source, i.e. Bt[n][k] row-major) uses the symmetric
// packing with n = lane&15.
// ---------------------------------------------------------------------------
__device__ __forceinline__ v16h load_frag(const _Float16* rowPtr, int kBase, int half) {
    v8h lo = *(const v8h*)(rowPtr + kBase + 8 * half);
    v8h hi = *(const v8h*)(rowPtr + kBase + 16 + 8 * half);
    v16h r;
#pragma unroll
    for (int i = 0; i < 8; ++i) { r[i] = lo[i]; r[i + 8] = hi[i]; }
    return r;
}

// ---------------------------------------------------------------------------
// Generic WMMA GEMM:  C[M,N] = act(A[M,K] * Bt[N,K]^T + bias)
// A, Bt are f16 row-major; C f32 row-major. One wave per 16x16 tile,
// 4 waves per block along N. M,N multiples of 16; K multiple of 32.
// act: 0 = none, 1 = tanh
// ---------------------------------------------------------------------------
__global__ void wmma_gemm(const _Float16* __restrict__ A,
                          const _Float16* __restrict__ Bt,
                          const float* __restrict__ bias,
                          float* __restrict__ C,
                          int M, int N, int K, int act) {
    int lane = threadIdx.x & 31;
    int wave = threadIdx.x >> 5;
    int tn   = blockIdx.x * 4 + wave;
    if (tn * 16 >= N) return;                   // wave-uniform exit (EXEC stays full)
    int tm   = blockIdx.y;
    int half = lane >> 4;
    int r16  = lane & 15;

    const _Float16* arow = A  + (size_t)(tm * 16 + r16) * K;
    const _Float16* brow = Bt + (size_t)(tn * 16 + r16) * K;

    v8f acc = {};
    for (int k = 0; k < K; k += 32) {
        v16h a = load_frag(arow, k, half);
        v16h b = load_frag(brow, k, half);
        acc = __builtin_amdgcn_wmma_f32_16x16x32_f16(false, a, false, b,
                                                     (short)0, acc, false, false);
    }
    int col  = tn * 16 + r16;
    int row0 = tm * 16 + half * 8;
    float bv = bias ? bias[col] : 0.0f;
#pragma unroll
    for (int r = 0; r < 8; ++r) {
        float v = acc[r] + bv;
        if (act == 1) v = tanhf(v);
        C[(size_t)(row0 + r) * N + col] = v;
    }
}

// ---------------------------------------------------------------------------
// Big logits GEMM: 16x64 per wave (A fragment reused 4x), prefetch on the
// streamed B (normalized embedding) rows. N need not be a multiple of 64.
// ---------------------------------------------------------------------------
__global__ void wmma_gemm_big(const _Float16* __restrict__ A,
                              const _Float16* __restrict__ Bt,
                              float* __restrict__ C,
                              int M, int N, int K) {
    int lane = threadIdx.x & 31;
    int wave = threadIdx.x >> 5;
    int tn0  = (blockIdx.x * 4 + wave) * 64;
    if (tn0 >= N) return;
    int tm   = blockIdx.y;
    int half = lane >> 4;
    int r16  = lane & 15;
    int jmax = (N - tn0) / 16; if (jmax > 4) jmax = 4;

    const _Float16* arow = A + (size_t)(tm * 16 + r16) * K;
    v8f acc[4] = {};
    for (int k = 0; k < K; k += 32) {
        v16h a = load_frag(arow, k, half);
        if (tn0 + 256 + r16 < N)   // prefetch 4 tiles ahead -> global_prefetch_b8
            __builtin_prefetch(Bt + (size_t)(tn0 + 256 + r16) * K + k, 0, 1);
#pragma unroll
        for (int j = 0; j < 4; ++j) {
            if (j < jmax) {
                const _Float16* brow = Bt + (size_t)(tn0 + j * 16 + r16) * K;
                v16h b = load_frag(brow, k, half);
                acc[j] = __builtin_amdgcn_wmma_f32_16x16x32_f16(false, a, false, b,
                                                                (short)0, acc[j], false, false);
            }
        }
    }
    int row0 = tm * 16 + half * 8;
#pragma unroll
    for (int j = 0; j < 4; ++j) {
        if (j < jmax) {
            int col = tn0 + j * 16 + r16;
#pragma unroll
            for (int r = 0; r < 8; ++r)
                C[(size_t)(row0 + r) * N + col] = acc[j][r];
        }
    }
}

// ---------------------------------------------------------------------------
// Transpose + f32->f16 convert:  dst[Nn,K] = (f16) src[K,Nn]^T
// ---------------------------------------------------------------------------
__global__ void tconv_kernel(const float* __restrict__ src, _Float16* __restrict__ dst,
                             int K, int Nn) {
    int n = blockIdx.x * blockDim.x + threadIdx.x;
    int k = blockIdx.y;
    if (n < Nn) dst[(size_t)n * K + k] = (_Float16)src[(size_t)k * Nn + n];
}

// ---------------------------------------------------------------------------
// Row L2-normalize (D=128), optional gather index, optional f32/f16 outputs
// with independent row strides. One 128-thread block per row.
// ---------------------------------------------------------------------------
__global__ void normalize_rows(const float* __restrict__ src, const int* __restrict__ idx,
                               int srcStride,
                               float* __restrict__ out32, int out32Stride,
                               _Float16* __restrict__ out16, int out16Stride) {
    int row = blockIdx.x;
    int t   = threadIdx.x;
    int sr  = idx ? idx[row] : row;
    float v = src[(size_t)sr * srcStride + t];
    __shared__ float sh[DD];
    sh[t] = v * v;
    __syncthreads();
    for (int off = 64; off > 0; off >>= 1) {
        if (t < off) sh[t] += sh[t + off];
        __syncthreads();
    }
    float nv = v / fmaxf(sqrtf(sh[0]), 1e-12f);
    if (out32) out32[(size_t)row * out32Stride + t] = nv;
    if (out16) out16[(size_t)row * out16Stride + t] = (_Float16)nv;
}

__global__ void fillf_kernel(float* __restrict__ p, float v, size_t n) {
    for (size_t i = (size_t)blockIdx.x * blockDim.x + threadIdx.x; i < n;
         i += (size_t)gridDim.x * blockDim.x)
        p[i] = v;
}

// ---------------------------------------------------------------------------
// GAT edge pipeline
// ---------------------------------------------------------------------------
__device__ __forceinline__ void atomicMaxF(float* addr, float val) {
    int* ai = reinterpret_cast<int*>(addr);
    int old = *ai;
    while (__int_as_float(old) < val) {
        int assumed = old;
        old = atomicCAS(ai, assumed, __float_as_int(val));
        if (old == assumed) break;
    }
}

// e = leaky_relu( sum_k ft[s][k]*ft[d][k]*pi[k] ), one wave per edge
__global__ void edge_score_kernel(const float* __restrict__ ft,
                                  const int* __restrict__ s, const int* __restrict__ d,
                                  const float* __restrict__ pi,
                                  float* __restrict__ esc, int E) {
    int e    = blockIdx.x * (blockDim.x >> 5) + (threadIdx.x >> 5);
    int lane = threadIdx.x & 31;
    if (e >= E) return;
    const float* fs = ft + (size_t)s[e] * HD;
    const float* fd = ft + (size_t)d[e] * HD;
    float p = 0.0f;
    for (int k = lane; k < HD; k += 32) p += fs[k] * fd[k] * pi[k];
#pragma unroll
    for (int off = 16; off > 0; off >>= 1) p += __shfl_down(p, off, 32);
    if (lane == 0) esc[e] = (p > 0.0f) ? p : 0.2f * p;
}

__global__ void edge_max_kernel(const float* __restrict__ esc, const int* __restrict__ d,
                                float* __restrict__ mmax, int E) {
    int e = blockIdx.x * blockDim.x + threadIdx.x;
    if (e < E) atomicMaxF(&mmax[d[e]], esc[e]);
}

__global__ void edge_expsum_kernel(float* __restrict__ esc, const int* __restrict__ d,
                                   const float* __restrict__ mmax, float* __restrict__ den,
                                   int E) {
    int e = blockIdx.x * blockDim.x + threadIdx.x;
    if (e < E) {
        float ex = expf(esc[e] - mmax[d[e]]);
        esc[e] = ex;
        atomicAdd(&den[d[e]], ex);
    }
}

// out[d] += ft[s] * a  (a = ex/den), one 256-thread block per edge
__global__ void edge_scatter_kernel(const float* __restrict__ ft,
                                    const int* __restrict__ s, const int* __restrict__ d,
                                    const float* __restrict__ esc, const float* __restrict__ den,
                                    float* __restrict__ gout) {
    int e = blockIdx.x;
    float a = esc[e] / den[d[e]];
    const float* fs = ft + (size_t)s[e] * HD;
    float* go = gout + (size_t)d[e] * HD;
    for (int k = threadIdx.x; k < HD; k += blockDim.x)
        atomicAdd(&go[k], fs[k] * a);
}

// h = normalize(2*h_v + headmax(go1) + headmax(go2));  also builds cat=[h,pos] f16
__global__ void headmax_kernel(const float* __restrict__ hv32,
                               const float* __restrict__ go1, const float* __restrict__ go2,
                               const float* __restrict__ pos_emb, const int* __restrict__ pid,
                               float* __restrict__ h32, _Float16* __restrict__ cat16) {
    int n = blockIdx.x;
    int t = threadIdx.x;
    float m1 = -3.4e38f, m2 = -3.4e38f;
#pragma unroll
    for (int h = 0; h < HH; ++h) {
        m1 = fmaxf(m1, go1[(size_t)n * HD + h * DD + t]);
        m2 = fmaxf(m2, go2[(size_t)n * HD + h * DD + t]);
    }
    float pre = 2.0f * hv32[(size_t)n * DD + t] + m1 + m2;
    __shared__ float sh[DD];
    sh[t] = pre * pre;
    __syncthreads();
    for (int off = 64; off > 0; off >>= 1) {
        if (t < off) sh[t] += sh[t + off];
        __syncthreads();
    }
    float nv = pre / fmaxf(sqrtf(sh[0]), 1e-12f);
    h32[(size_t)n * DD + t] = nv;
    cat16[(size_t)n * 256 + t]      = (_Float16)nv;
    cat16[(size_t)n * 256 + DD + t] = (_Float16)pos_emb[(size_t)pid[n] * DD + t];
}

// coef[n] = <eq[n], tgt_emb[tid[agg[n]]]>
__global__ void coef_kernel(const float* __restrict__ eq, const float* __restrict__ tgt_emb,
                            const int* __restrict__ agg, const int* __restrict__ tid,
                            float* __restrict__ coef) {
    int n = blockIdx.x;
    int t = threadIdx.x;
    const float* tr = tgt_emb + (size_t)tid[agg[n]] * DD;
    __shared__ float sh[DD];
    sh[t] = eq[(size_t)n * DD + t] * tr[t];
    __syncthreads();
    for (int off = 64; off > 0; off >>= 1) {
        if (t < off) sh[t] += sh[t + off];
        __syncthreads();
    }
    if (t == 0) coef[n] = sh[0];
}

// Per-session: srcat[b] = [ h[last_idx[b]] | sum_j h[b*SLEN+j]*coef ]  (f16)
__global__ void session_kernel(const float* __restrict__ h32, const float* __restrict__ coef,
                               const int* __restrict__ lastx, _Float16* __restrict__ srcat) {
    int b = blockIdx.x;
    int t = threadIdx.x;
    srcat[(size_t)b * 256 + t] = (_Float16)h32[(size_t)lastx[b] * DD + t];
    float acc = 0.0f;
#pragma unroll
    for (int j = 0; j < SLEN; ++j) {
        int n = b * SLEN + j;
        acc += h32[(size_t)n * DD + t] * coef[n];
    }
    srcat[(size_t)b * 256 + DD + t] = (_Float16)acc;
}

// ---------------------------------------------------------------------------
// Online softmax row stats over V, one 256-thread block per row (b,o)
// ---------------------------------------------------------------------------
__global__ void row_stats_kernel(const float* __restrict__ logits,
                                 float* __restrict__ rowm, float* __restrict__ rowd, int V) {
    int bo = blockIdx.x;
    int t  = threadIdx.x;
    const float* lp = logits + (size_t)bo * V;
    float m = -3.4e38f, s = 0.0f;
    for (int v = t; v < V; v += blockDim.x) {
        float x = 12.0f * lp[v];
        if (x > m) { s = s * expf(m - x) + 1.0f; m = x; }
        else       { s += expf(x - m); }
    }
    __shared__ float sm[256], ss[256];
    sm[t] = m; ss[t] = s;
    __syncthreads();
    for (int off = 128; off > 0; off >>= 1) {
        if (t < off) {
            float m2 = sm[t + off], s2 = ss[t + off];
            float M  = fmaxf(sm[t], m2);
            ss[t] = ss[t] * expf(sm[t] - M) + s2 * expf(m2 - M);
            sm[t] = M;
        }
        __syncthreads();
    }
    if (t == 0) { rowm[bo] = sm[0]; rowd[bo] = ss[0]; }
}

// out[b,v] = log( sum_o softmax(alpha)[o] * exp(12*l[bo,v]-m[bo]) / den[bo] )
__global__ void final_kernel(const float* __restrict__ logits,
                             const float* __restrict__ rowm, const float* __restrict__ rowd,
                             const float* __restrict__ alpha, float* __restrict__ out,
                             int V, int B) {
    size_t i = (size_t)blockIdx.x * blockDim.x + threadIdx.x;
    if (i >= (size_t)B * V) return;
    int b = (int)(i / V);
    int v = (int)(i % V);
    float a0 = alpha[0], a1 = alpha[1], a2 = alpha[2];
    float am = fmaxf(a0, fmaxf(a1, a2));
    float e0 = expf(a0 - am), e1 = expf(a1 - am), e2 = expf(a2 - am);
    float inv = 1.0f / (e0 + e1 + e2);
    float al[3] = { e0 * inv, e1 * inv, e2 * inv };
    float s = 0.0f;
#pragma unroll
    for (int o = 0; o < OO; ++o) {
        int bo = b * OO + o;
        s += al[o] * expf(12.0f * logits[(size_t)bo * V + v] - rowm[bo]) / rowd[bo];
    }
    out[i] = logf(s);
}

// ---------------------------------------------------------------------------
// Host orchestration
// ---------------------------------------------------------------------------
extern "C" void kernel_launch(void* const* d_in, const int* in_sizes, int n_in,
                              void* d_out, int out_size, void* d_ws, size_t ws_size,
                              hipStream_t stream) {
    const float* embedding = (const float*)d_in[0];
    const float* pos_emb   = (const float*)d_in[1];
    const float* tgt_emb   = (const float*)d_in[2];
    const float* g1w  = (const float*)d_in[3];
    const float* g1b  = (const float*)d_in[4];
    const float* g1p  = (const float*)d_in[5];
    const float* g2w  = (const float*)d_in[6];
    const float* g2b  = (const float*)d_in[7];
    const float* g2p  = (const float*)d_in[8];
    const float* qw   = (const float*)d_in[9];
    const float* srw  = (const float*)d_in[10];
    const float* alpha = (const float*)d_in[11];
    const int* iid   = (const int*)d_in[12];
    const int* srcA  = (const int*)d_in[13];
    const int* dstA  = (const int*)d_in[14];
    const int* agg   = (const int*)d_in[15];
    const int* pid   = (const int*)d_in[16];
    const int* tid   = (const int*)d_in[17];
    const int* lastx = (const int*)d_in[18];
    float* out = (float*)d_out;

    // --- workspace bump allocator (total ~435 MB; logits buffer dominates) ---
    char* wp = (char*)d_ws;
    auto alloc = [&](size_t bytes) -> void* {
        void* r = (void*)wp;
        wp += (bytes + 255) & ~(size_t)255;
        return r;
    };
    _Float16* bb16  = (_Float16*)alloc((size_t)VV * DD * 2);
    float*    hv32  = (float*)   alloc((size_t)NN * DD * 4);
    _Float16* hv16  = (_Float16*)alloc((size_t)NN * DD * 2);
    _Float16* wt1   = (_Float16*)alloc((size_t)OO * HD * DD * 2);
    _Float16* wt2   = (_Float16*)alloc((size_t)OO * HD * DD * 2);
    _Float16* qT    = (_Float16*)alloc((size_t)OO * DD * 256 * 2);
    _Float16* srT   = (_Float16*)alloc((size_t)OO * DD * 256 * 2);
    float*    ft1   = (float*)   alloc((size_t)NN * HD * 4);
    float*    ft2   = (float*)   alloc((size_t)NN * HD * 4);
    float*    go1   = (float*)   alloc((size_t)NN * HD * 4);
    float*    go2   = (float*)   alloc((size_t)NN * HD * 4);
    float*    esc   = (float*)   alloc((size_t)EE * 4);
    float*    mmax  = (float*)   alloc((size_t)NN * 4);
    float*    den   = (float*)   alloc((size_t)NN * 4);
    float*    h32   = (float*)   alloc((size_t)NN * DD * 4);
    _Float16* cat16 = (_Float16*)alloc((size_t)NN * 256 * 2);
    float*    eq    = (float*)   alloc((size_t)NN * DD * 4);
    float*    coef  = (float*)   alloc((size_t)NN * 4);
    _Float16* srcat = (_Float16*)alloc((size_t)BB * 256 * 2);
    float*    srtmp = (float*)   alloc((size_t)BB * DD * 4);
    _Float16* sr16  = (_Float16*)alloc((size_t)BB * OO * DD * 2);
    float*    logits= (float*)   alloc((size_t)BB * OO * VV * 4);
    float*    rowm  = (float*)   alloc((size_t)BB * OO * 4);
    float*    rowd  = (float*)   alloc((size_t)BB * OO * 4);

    // --- weight transpose/convert to f16 (Bt layout [Nout, K]) ---
    for (int o = 0; o < OO; ++o) {
        tconv_kernel<<<dim3((HD + 255) / 256, DD), 256, 0, stream>>>(
            g1w + (size_t)o * DD * HD, wt1 + (size_t)o * HD * DD, DD, HD);
        tconv_kernel<<<dim3((HD + 255) / 256, DD), 256, 0, stream>>>(
            g2w + (size_t)o * DD * HD, wt2 + (size_t)o * HD * DD, DD, HD);
        tconv_kernel<<<dim3(1, 256), 256, 0, stream>>>(
            qw + (size_t)o * 256 * DD, qT + (size_t)o * DD * 256, 256, DD);
        tconv_kernel<<<dim3(1, 256), 256, 0, stream>>>(
            srw + (size_t)o * 256 * DD, srT + (size_t)o * DD * 256, 256, DD);
    }

    // --- bb = normalize(embedding) -> f16 ; h_v = normalize(embedding[iid]) ---
    normalize_rows<<<VV, DD, 0, stream>>>(embedding, nullptr, DD,
                                          nullptr, 0, bb16, DD);
    normalize_rows<<<NN, DD, 0, stream>>>(embedding, iid, DD,
                                          hv32, DD, hv16, DD);

    for (int o = 0; o < OO; ++o) {
        const int* so = srcA + (size_t)o * EE;
        const int* doo = dstA + (size_t)o * EE;

        // ft = h_v @ W + b  (two layers)
        dim3 gft((HD / 16 + 3) / 4, NN / 16);
        wmma_gemm<<<gft, 128, 0, stream>>>(hv16, wt1 + (size_t)o * HD * DD,
                                           g1b + (size_t)o * HD, ft1, NN, HD, DD, 0);
        wmma_gemm<<<gft, 128, 0, stream>>>(hv16, wt2 + (size_t)o * HD * DD,
                                           g2b + (size_t)o * HD, ft2, NN, HD, DD, 0);

        // layer 1: edges (src -> dst), layer 2: reversed graph (dst -> src)
        for (int l = 0; l < 2; ++l) {
            const float* ft  = (l == 0) ? ft1 : ft2;
            const float* pi  = (l == 0) ? g1p + (size_t)o * HD : g2p + (size_t)o * HD;
            float*       go  = (l == 0) ? go1 : go2;
            const int*   es  = (l == 0) ? so  : doo;
            const int*   ed  = (l == 0) ? doo : so;

            fillf_kernel<<<64, 256, 0, stream>>>(mmax, -3.4e38f, (size_t)NN);
            fillf_kernel<<<64, 256, 0, stream>>>(den, 0.0f, (size_t)NN);
            fillf_kernel<<<4096, 256, 0, stream>>>(go, 0.0f, (size_t)NN * HD);

            edge_score_kernel<<<(EE + 7) / 8, 256, 0, stream>>>(ft, es, ed, pi, esc, EE);
            edge_max_kernel<<<(EE + 255) / 256, 256, 0, stream>>>(esc, ed, mmax, EE);
            edge_expsum_kernel<<<(EE + 255) / 256, 256, 0, stream>>>(esc, ed, mmax, den, EE);
            edge_scatter_kernel<<<EE, 256, 0, stream>>>(ft, es, ed, esc, den, go);
        }

        // h = normalize(2*h_v + headmax1 + headmax2); build [h|pos] f16
        headmax_kernel<<<NN, DD, 0, stream>>>(hv32, go1, go2, pos_emb, pid, h32, cat16);

        // e = tanh([h|pos] @ q_w[o])
        dim3 gq((DD / 16 + 3) / 4, NN / 16);
        wmma_gemm<<<gq, 128, 0, stream>>>(cat16, qT + (size_t)o * DD * 256,
                                          nullptr, eq, NN, DD, 256, 1);

        coef_kernel<<<NN, DD, 0, stream>>>(eq, tgt_emb, agg, tid, coef);
        session_kernel<<<BB, DD, 0, stream>>>(h32, coef, lastx, srcat);

        // sr_o = normalize([last|feat] @ fc_sr_w[o]) -> sr16 rows b*O+o
        dim3 gsr((DD / 16 + 3) / 4, BB / 16);
        wmma_gemm<<<gsr, 128, 0, stream>>>(srcat, srT + (size_t)o * DD * 256,
                                           nullptr, srtmp, BB, DD, 256, 0);
        normalize_rows<<<BB, DD, 0, stream>>>(srtmp, nullptr, DD,
                                              nullptr, 0, sr16 + o * DD, OO * DD);
    }

    // logits[bo, v] = sr16[bo] . bb16[v]   (768 x 100000, K=128)
    int waveTiles = (VV + 63) / 64;
    dim3 gbig((waveTiles + 3) / 4, (BB * OO) / 16);
    wmma_gemm_big<<<gbig, 128, 0, stream>>>(sr16, bb16, logits, BB * OO, VV, DD);

    // softmax stats + fused final log-mixture
    row_stats_kernel<<<BB * OO, 256, 0, stream>>>(logits, rowm, rowd, VV);
    size_t tot = (size_t)BB * VV;
    final_kernel<<<(int)((tot + 255) / 256), 256, 0, stream>>>(logits, rowm, rowd,
                                                               alpha, out, VV, BB);
}
